// Net_71073118814859
// MI455X (gfx1250) — compile-verified, matrix operand order checked by reading, not run on hardware
//
#include <hip/hip_runtime.h>
#include <hip/hip_bf16.h>
#include <math.h>

#define N_NODES 100000
#define F_DIM   256
#define H_DIM   256
#define C_DIM   40
#define E_EDGES 3200000
#define K_HOPS  10
#define SLOTS   (K_HOPS + 1)

typedef __attribute__((ext_vector_type(16))) __bf16 v16bf;
typedef __attribute__((ext_vector_type(8)))  float  v8f;

union BfFrag {
    v16bf v;
    uint4 q[2];
    unsigned short h[16];
};

__device__ __forceinline__ unsigned short f2bf_bits(float f) {
    unsigned u = __float_as_uint(f);
    unsigned r = u + 0x7FFFu + ((u >> 16) & 1u);   // round-to-nearest-even
    return (unsigned short)(r >> 16);
}

// ---------------------------------------------------------------------------
// Prep: convert W1 [256x256] and W2 [256x40] (f32, row-major) into bf16
// B-fragment layout for v_wmma_f32_16x16x32_bf16:
//   fragment (jt, kb): 32 lanes x 16 halves, lane L holds column jt*16+(L&15),
//   K-slices kb*32 + koff + {0..7, 16..23} with koff = (L>=16)?8:0.
// Storage: W1B[((jt*8+kb)*32 + lane)*16 + e]  -> one 32B load per fragment.
// ---------------------------------------------------------------------------
__global__ void prep_weights(const float* __restrict__ W1,
                             const float* __restrict__ W2,
                             unsigned short* __restrict__ W1B,
                             unsigned short* __restrict__ W2B) {
    int t = blockIdx.x * blockDim.x + threadIdx.x;
    const int n1 = 16 * 8 * 32 * 16;   // W1 fragments
    const int n2 = 3 * 8 * 32 * 16;    // W2 fragments (C padded to 48)
    if (t < n1) {
        int e = t & 15, lane = (t >> 4) & 31, kb = (t >> 9) & 7, jt = t >> 12;
        int col  = jt * 16 + (lane & 15);
        int koff = (lane >= 16) ? 8 : 0;
        int k    = kb * 32 + koff + ((e < 8) ? e : (8 + e));
        W1B[t] = f2bf_bits(W1[k * H_DIM + col]);
    } else if (t < n1 + n2) {
        int t2 = t - n1;
        int e = t2 & 15, lane = (t2 >> 4) & 31, kb = (t2 >> 9) & 7, jt = t2 >> 12;
        int col  = jt * 16 + (lane & 15);
        int koff = (lane >= 16) ? 8 : 0;
        int k    = kb * 32 + koff + ((e < 8) ? e : (8 + e));
        float v  = (col < C_DIM) ? W2[k * C_DIM + col] : 0.0f;
        W2B[t2] = f2bf_bits(v);
    }
}

// ---------------------------------------------------------------------------
// Fused MLP: one wave (32 threads) per 16-row node tile.
//   h  = relu(x @ W1 + b1)      via 16 x 8 WMMA bf16
//   h2 = h @ W2 + b2            via 3 x 8 WMMA bf16  -> pps[:, slot 0, :]
// ---------------------------------------------------------------------------
__global__ void __launch_bounds__(32) mlp_kernel(
    const float* __restrict__ x,  const float* __restrict__ b1,
    const float* __restrict__ b2,
    const unsigned short* __restrict__ W1B,
    const unsigned short* __restrict__ W2B,
    float* __restrict__ pps)
{
    __shared__ __align__(16) unsigned short hbuf[16 * H_DIM];   // 8 KB bf16 tile

    const int lane = threadIdx.x;
    const int row0 = blockIdx.x * 16;
    const int m15  = lane & 15;
    const int hi   = (lane >= 16) ? 1 : 0;

    // --- A fragments of x (lane m15 holds row m15) ---
    BfFrag a[8];
    const float* xrow = x + (size_t)(row0 + m15) * F_DIM;
    #pragma unroll
    for (int kb = 0; kb < 8; ++kb) {
        const int ks = kb * 32 + hi * 8;
        float4 u0 = *(const float4*)(xrow + ks);
        float4 u1 = *(const float4*)(xrow + ks + 4);
        float4 w0 = *(const float4*)(xrow + ks + 16);
        float4 w1 = *(const float4*)(xrow + ks + 20);
        a[kb].h[0]  = f2bf_bits(u0.x); a[kb].h[1]  = f2bf_bits(u0.y);
        a[kb].h[2]  = f2bf_bits(u0.z); a[kb].h[3]  = f2bf_bits(u0.w);
        a[kb].h[4]  = f2bf_bits(u1.x); a[kb].h[5]  = f2bf_bits(u1.y);
        a[kb].h[6]  = f2bf_bits(u1.z); a[kb].h[7]  = f2bf_bits(u1.w);
        a[kb].h[8]  = f2bf_bits(w0.x); a[kb].h[9]  = f2bf_bits(w0.y);
        a[kb].h[10] = f2bf_bits(w0.z); a[kb].h[11] = f2bf_bits(w0.w);
        a[kb].h[12] = f2bf_bits(w1.x); a[kb].h[13] = f2bf_bits(w1.y);
        a[kb].h[14] = f2bf_bits(w1.z); a[kb].h[15] = f2bf_bits(w1.w);
    }

    // --- GEMM1 + bias + ReLU, stash bf16 h-tile in LDS (D->A relayout) ---
    for (int jt = 0; jt < 16; ++jt) {
        v8f acc = {0.f, 0.f, 0.f, 0.f, 0.f, 0.f, 0.f, 0.f};
        #pragma unroll
        for (int kb = 0; kb < 8; ++kb) {
            BfFrag b;
            const uint4* bp = (const uint4*)(W1B + ((size_t)(jt * 8 + kb) * 32 + lane) * 16);
            b.q[0] = bp[0]; b.q[1] = bp[1];
            acc = __builtin_amdgcn_wmma_f32_16x16x32_bf16(
                      false, a[kb].v, false, b.v, (short)0, acc, false, false);
        }
        const float bias = b1[jt * 16 + m15];
        #pragma unroll
        for (int r = 0; r < 8; ++r) {                 // D layout: row r(+8), col m15
            float v = acc[r] + bias;
            v = v > 0.f ? v : 0.f;
            hbuf[(r + hi * 8) * H_DIM + jt * 16 + m15] = f2bf_bits(v);
        }
    }
    __syncthreads();

    // --- GEMM2: h (A from LDS) @ W2 -> pps slot 0 ---
    #pragma unroll
    for (int jt = 0; jt < 3; ++jt) {
        v8f acc = {0.f, 0.f, 0.f, 0.f, 0.f, 0.f, 0.f, 0.f};
        #pragma unroll
        for (int kb = 0; kb < 8; ++kb) {
            const int ks = kb * 32 + hi * 8;
            BfFrag a2;
            a2.q[0] = *(const uint4*)(hbuf + m15 * H_DIM + ks);
            a2.q[1] = *(const uint4*)(hbuf + m15 * H_DIM + ks + 16);
            BfFrag b;
            const uint4* bp = (const uint4*)(W2B + ((size_t)(jt * 8 + kb) * 32 + lane) * 16);
            b.q[0] = bp[0]; b.q[1] = bp[1];
            acc = __builtin_amdgcn_wmma_f32_16x16x32_bf16(
                      false, a2.v, false, b.v, (short)0, acc, false, false);
        }
        const int c = jt * 16 + m15;
        if (c < C_DIM) {
            const float bias = b2[c];
            #pragma unroll
            for (int r = 0; r < 8; ++r) {
                const int m = r + hi * 8;
                pps[((size_t)(row0 + m) * SLOTS + 0) * C_DIM + c] = acc[r] + bias;
            }
        }
    }
}

// ---------------------------------------------------------------------------
// One propagation hop: thread per (edge, channel); coalesced gather + f32
// L2 atomics. pps slots are 16MB each -> L2 resident on 192MB L2.
// ---------------------------------------------------------------------------
__global__ void hop_kernel(const int* __restrict__ ei,
                           const float* __restrict__ norm,
                           float* __restrict__ pps, int slot)
{
    unsigned tid = blockIdx.x * blockDim.x + threadIdx.x;
    if (tid >= (unsigned)E_EDGES * C_DIM) return;
    unsigned e = tid / C_DIM;
    unsigned c = tid - e * C_DIM;
    int   src = ei[e];
    int   dst = ei[E_EDGES + e];
    float w   = norm[e];
    float val = w * pps[((size_t)src * SLOTS + (slot - 1)) * C_DIM + c];
    atomicAdd(&pps[((size_t)dst * SLOTS + slot) * C_DIM + c], val);
}

// ---------------------------------------------------------------------------
// Readout: one wave32 per node. score_k = sigmoid(pps[n,k,:]·Wp + bp),
// out = sum_k score_k * pps[n,k,:], then log_softmax over 40 classes.
// ---------------------------------------------------------------------------
__global__ void __launch_bounds__(256) final_kernel(
    const float* __restrict__ pps, const float* __restrict__ Wp,
    const float* __restrict__ bp, float* __restrict__ out)
{
    const int wid  = (int)((blockIdx.x * blockDim.x + threadIdx.x) >> 5);
    const int lane = threadIdx.x & 31;
    if (wid >= N_NODES) return;

    const int  c0   = lane;
    const int  c1   = lane + 32;
    const bool has1 = (c1 < C_DIM);

    const float wp0 = Wp[c0];
    const float wp1 = has1 ? Wp[c1] : 0.f;
    const float bpv = bp[0];

    float out0 = 0.f, out1 = 0.f;
    const float* base = pps + (size_t)wid * SLOTS * C_DIM;

    #pragma unroll
    for (int k = 0; k < SLOTS; ++k) {
        float p0 = base[k * C_DIM + c0];
        float p1 = has1 ? base[k * C_DIM + c1] : 0.f;
        float part = p0 * wp0 + p1 * wp1;
        #pragma unroll
        for (int off = 16; off >= 1; off >>= 1)
            part += __shfl_xor(part, off, 32);
        float s = 1.f / (1.f + expf(-(part + bpv)));
        out0 += s * p0;
        out1 += s * p1;
    }

    float m = fmaxf(out0, has1 ? out1 : -INFINITY);
    #pragma unroll
    for (int off = 16; off >= 1; off >>= 1)
        m = fmaxf(m, __shfl_xor(m, off, 32));
    float se = expf(out0 - m) + (has1 ? expf(out1 - m) : 0.f);
    #pragma unroll
    for (int off = 16; off >= 1; off >>= 1)
        se += __shfl_xor(se, off, 32);
    float lse = m + logf(se);

    out[(size_t)wid * C_DIM + c0] = out0 - lse;
    if (has1) out[(size_t)wid * C_DIM + c1] = out1 - lse;
}

// ---------------------------------------------------------------------------
extern "C" void kernel_launch(void* const* d_in, const int* in_sizes, int n_in,
                              void* d_out, int out_size, void* d_ws, size_t ws_size,
                              hipStream_t stream) {
    (void)in_sizes; (void)n_in; (void)out_size; (void)ws_size;

    const float* x    = (const float*)d_in[0];
    const int*   ei   = (const int*)  d_in[1];
    const float* norm = (const float*)d_in[2];
    const float* W1   = (const float*)d_in[3];
    const float* b1   = (const float*)d_in[4];
    const float* W2   = (const float*)d_in[5];
    const float* b2   = (const float*)d_in[6];
    const float* Wp   = (const float*)d_in[7];
    const float* bp   = (const float*)d_in[8];
    float*       out  = (float*)d_out;

    // Workspace layout: W1B (128 KB) | W2B (24 KB, pad to 0x26000) | pps (176 MB)
    unsigned short* W1B = (unsigned short*)d_ws;
    unsigned short* W2B = (unsigned short*)((char*)d_ws + 131072);
    float*          pps = (float*)((char*)d_ws + 155648);

    // pps must be zeroed every call (slots 1..10 are atomically accumulated)
    hipMemsetAsync(pps, 0, (size_t)N_NODES * SLOTS * C_DIM * sizeof(float), stream);

    const int prep_threads = 16 * 8 * 32 * 16 + 3 * 8 * 32 * 16;
    prep_weights<<<(prep_threads + 255) / 256, 256, 0, stream>>>(W1, W2, W1B, W2B);

    mlp_kernel<<<N_NODES / 16, 32, 0, stream>>>(x, b1, b2, W1B, W2B, pps);

    const unsigned hop_threads = (unsigned)E_EDGES * C_DIM;
    for (int k = 1; k <= K_HOPS; ++k)
        hop_kernel<<<(hop_threads + 255) / 256, 256, 0, stream>>>(ei, norm, pps, k);

    final_kernel<<<(N_NODES * 32 + 255) / 256, 256, 0, stream>>>(pps, Wp, bp, out);
}
